// batch_BiLSTM_CRF_70781061038243
// MI455X (gfx1250) — compile-verified, matrix operand order checked by reading, not run on hardware
//
#include <hip/hip_runtime.h>
#include <hip/hip_bf16.h>
#include <stdint.h>

// ---------------------------------------------------------------- types
typedef __attribute__((ext_vector_type(16))) __bf16 v16bf;
typedef __attribute__((ext_vector_type(8)))  float  v8f;

#define E_DIM   256
#define HID     512
#define T_TAGS  12
#define H_DIM   256          // per-direction hidden
#define B_DIM   64
#define S_DIM   256
#define G4H     1024         // 4*H
#define START_TAG 10
#define STOP_TAG  11
#define NEGV (-10000.0f)

// ---------------------------------------------------------------- helpers
__device__ __forceinline__ unsigned short f2bf(float f) {
  unsigned int u = __builtin_bit_cast(unsigned int, f);
  u += 0x7FFFu + ((u >> 16) & 1u);          // round-to-nearest-even
  return (unsigned short)(u >> 16);
}
__device__ __forceinline__ float bf2f(unsigned short s) {
  unsigned int u = ((unsigned int)s) << 16;
  return __builtin_bit_cast(float, u);
}
__device__ __forceinline__ __bf16 us2b(unsigned short s) {
  return __builtin_bit_cast(__bf16, s);
}
__device__ __forceinline__ float sigf(float x) { return 1.0f / (1.0f + expf(-x)); }

// Load a 16x32 bf16 fragment (A-layout per ISA 7.12.2; B uses the mirrored
// layout with N striped across lanes, which is the same index math when the
// source is the row-major weight matrix W[N][K], i.e. B = W^T tile).
__device__ __forceinline__ v16bf load_frag16x32(const unsigned short* base,
                                                int row0, int ld, int k0) {
  int lane = threadIdx.x & 31;
  int m    = lane & 15;
  int half = lane >> 4;
  const unsigned short* p = base + (size_t)(row0 + m) * ld + k0;
  v16bf r;
#pragma unroll
  for (int v = 0; v < 8; ++v) {
    int kb = ((v & 4) << 2) + half * 8 + ((v & 3) << 1);   // K offset pattern
    unsigned int pr = *(const unsigned int*)(p + kb);       // two bf16
    r[2 * v]     = us2b((unsigned short)(pr & 0xFFFFu));
    r[2 * v + 1] = us2b((unsigned short)(pr >> 16));
  }
  return r;
}

__device__ __forceinline__ v8f wmma_bf16(v16bf a, v16bf b, v8f c) {
  return __builtin_amdgcn_wmma_f32_16x16x32_bf16(
      /*neg_a=*/false, a, /*neg_b=*/false, b,
      /*c_mod=*/(short)0, c, /*reuse_a=*/false, /*reuse_b=*/false);
}

// ---------------------------------------------------------------- kernels

// Convert the 4 LSTM weight matrices (each 1024 x 256 f32) to bf16.
__global__ void prep_weights(const float* wihf, const float* whhf,
                             const float* wihb, const float* whhb,
                             unsigned short* owihf, unsigned short* owhhf,
                             unsigned short* owihb, unsigned short* owhhb) {
  int mat = blockIdx.x >> 10;
  int row = blockIdx.x & 1023;
  int col = threadIdx.x;
  const float* src = (mat == 0) ? wihf : (mat == 1) ? whhf : (mat == 2) ? wihb : whhb;
  unsigned short* dst = (mat == 0) ? owihf : (mat == 1) ? owhhf : (mat == 2) ? owihb : owhhb;
  size_t i = (size_t)row * E_DIM + col;
  dst[i] = f2bf(src[i]);
}

// bias_dir[n] = bih[n] + bhh[n]
__global__ void prep_bias(const float* bihf, const float* bhhf,
                          const float* bihb, const float* bhhb,
                          float* bsum_f, float* bsum_b) {
  int i = blockIdx.x * 256 + threadIdx.x;   // 0..2047
  if (i < G4H) bsum_f[i] = bihf[i] + bhhf[i];
  else         bsum_b[i - G4H] = bihb[i - G4H] + bhhb[i - G4H];
}

// x[s,b,:] = bf16(emb[sent[b,s]])  -> Xg row m = s*B + b
__global__ void gather_embed(const int* sent, const float* emb,
                             unsigned short* xg) {
  int m = blockIdx.x;
  int s = m >> 6, b = m & 63;
  int tok = sent[b * S_DIM + s];
  int e = threadIdx.x;
  xg[(size_t)m * E_DIM + e] = f2bf(emb[(size_t)tok * E_DIM + e]);
}

// G_dir = Xg @ Wih_dir^T + (bih+bhh)   [16384 x 1024], stored bf16
__global__ void input_proj(const unsigned short* xg,
                           const unsigned short* wih_f, const unsigned short* wih_b,
                           const float* bias_f, const float* bias_b,
                           unsigned short* gf, unsigned short* gb) {
  int wid  = threadIdx.x >> 5;
  int lane = threadIdx.x & 31;
  int dir  = blockIdx.y >> 4;
  int n0   = (blockIdx.y & 15) * 64;
  int m0   = blockIdx.x * 128 + wid * 16;
  const unsigned short* W = dir ? wih_b : wih_f;
  const float* bias = dir ? bias_b : bias_f;
  unsigned short* out = dir ? gb : gf;

  v8f acc[4] = {};
  for (int k0 = 0; k0 < E_DIM; k0 += 32) {
    v16bf a = load_frag16x32(xg, m0, E_DIM, k0);
#pragma unroll
    for (int jn = 0; jn < 4; ++jn) {
      v16bf bfr = load_frag16x32(W, n0 + 16 * jn, E_DIM, k0);
      acc[jn] = wmma_bf16(a, bfr, acc[jn]);
    }
  }
  int half = lane >> 4, nl = lane & 15;
#pragma unroll
  for (int jn = 0; jn < 4; ++jn) {
    int n = n0 + 16 * jn + nl;
    float bv = bias[n];
#pragma unroll
    for (int r = 0; r < 8; ++r) {
      int m = m0 + r + 8 * half;
      out[(size_t)m * G4H + n] = f2bf(acc[jn][r] + bv);
    }
  }
}

// Sequential LSTM scan, one workgroup (32 waves) per direction.
// h double-buffered in LDS (bf16); c in WMMA-layout registers.
__global__ void lstm_scan(const unsigned short* gf, const unsigned short* gb,
                          const unsigned short* whh_f, const unsigned short* whh_b,
                          const float* h0, const float* c0,
                          float* hf, float* hb) {
  __shared__ unsigned short hbuf[2][B_DIM * H_DIM];   // 2 x 32 KB
  int dir = blockIdx.x;
  const unsigned short* G = dir ? gb : gf;
  const unsigned short* W = dir ? whh_b : whh_f;
  float* hs = dir ? hb : hf;

  int tid = threadIdx.x;
  int wid = tid >> 5, lane = tid & 31;
  int half = lane >> 4, l15 = lane & 15;
  int m0  = (wid & 3) * 16;          // batch-row tile
  int hc0 = (wid >> 2) * 32;         // two h-column tiles: hc0, hc0+16

  // c state in registers (C/D layout)
  v8f creg[2];
#pragma unroll
  for (int j = 0; j < 2; ++j)
#pragma unroll
    for (int r = 0; r < 8; ++r) {
      int b = m0 + r + 8 * half;
      int h = hc0 + 16 * j + l15;
      creg[j][r] = c0[(size_t)dir * B_DIM * H_DIM + (size_t)b * H_DIM + h];
    }
  // h0 -> LDS buffer 0
  for (int i = tid; i < B_DIM * H_DIM; i += 1024)
    hbuf[0][i] = f2bf(h0[(size_t)dir * B_DIM * H_DIM + i]);
  __syncthreads();

  for (int js = 0; js < S_DIM; ++js) {
    int t   = dir ? (S_DIM - 1 - js) : js;
    int cur = js & 1;
    const unsigned short* Gt = G + (size_t)t * B_DIM * G4H;

    // accumulators start from precomputed input projection + bias
    v8f acc[2][4];
#pragma unroll
    for (int j = 0; j < 2; ++j)
#pragma unroll
      for (int g = 0; g < 4; ++g)
#pragma unroll
        for (int r = 0; r < 8; ++r) {
          int b = m0 + r + 8 * half;
          int n = g * H_DIM + hc0 + 16 * j + l15;
          acc[j][g][r] = bf2f(Gt[(size_t)b * G4H + n]);
        }

    // gates += h_prev @ Whh^T
    for (int k0 = 0; k0 < H_DIM; k0 += 32) {
      v16bf a = load_frag16x32(&hbuf[cur][0], m0, H_DIM, k0);
#pragma unroll
      for (int j = 0; j < 2; ++j)
#pragma unroll
        for (int g = 0; g < 4; ++g) {
          v16bf bb = load_frag16x32(W, g * H_DIM + hc0 + 16 * j, H_DIM, k0);
          acc[j][g] = wmma_bf16(a, bb, acc[j][g]);
        }
    }

    // prefetch next step's G rows for this wave (L2 warm-up)
    if (js + 1 < S_DIM) {
      int tn = dir ? (S_DIM - 2 - js) : (js + 1);
      __builtin_prefetch(G + (size_t)tn * B_DIM * G4H +
                             (size_t)(m0 + l15) * G4H + hc0, 0, 1);
    }

    // elementwise LSTM cell update entirely in registers
#pragma unroll
    for (int j = 0; j < 2; ++j)
#pragma unroll
      for (int r = 0; r < 8; ++r) {
        int b = m0 + r + 8 * half;
        int h = hc0 + 16 * j + l15;
        float ig = sigf(acc[j][0][r]);
        float fg = sigf(acc[j][1][r]);
        float gg = tanhf(acc[j][2][r]);
        float og = sigf(acc[j][3][r]);
        float c  = fg * creg[j][r] + ig * gg;
        creg[j][r] = c;
        float hv = og * tanhf(c);
        hs[(size_t)t * B_DIM * H_DIM + (size_t)b * H_DIM + h] = hv;
        hbuf[1 - cur][b * H_DIM + h] = f2bf(hv);
      }
    __syncthreads();
  }
}

// feats = concat(hf,hb) @ Wt^T + bt, then Viterbi decode. One block per b.
__global__ void feats_viterbi(const float* hf, const float* hb,
                              const float* Wt, const float* bt,
                              const float* trans, float* out) {
  __shared__ float wt_l[T_TAGS * HID];          // 24 KB
  __shared__ float bt_l[T_TAGS];
  __shared__ float tr_l[T_TAGS * T_TAGS];
  __shared__ float feats_l[S_DIM * T_TAGS];     // 12 KB
  __shared__ short bptr_l[S_DIM * T_TAGS];      // 6 KB
  __shared__ float fvbuf[2][T_TAGS];

  int b = blockIdx.x, tid = threadIdx.x;
  for (int i = tid; i < T_TAGS * HID; i += 256) wt_l[i] = Wt[i];
  if (tid < T_TAGS) bt_l[tid] = bt[tid];
  for (int i = tid; i < T_TAGS * T_TAGS; i += 256) tr_l[i] = trans[i];
  __syncthreads();

  // emissions: thread tid handles timestep s = tid
  {
    int s = tid;
    const float* hfp = hf + ((size_t)s * B_DIM + b) * H_DIM;
    const float* hbp = hb + ((size_t)s * B_DIM + b) * H_DIM;
    for (int tg = 0; tg < T_TAGS; ++tg) {
      float acc = bt_l[tg];
      const float* w = wt_l + tg * HID;
      for (int h = 0; h < H_DIM; ++h) acc += hfp[h] * w[h];
      for (int h = 0; h < H_DIM; ++h) acc += hbp[h] * w[H_DIM + h];
      feats_l[s * T_TAGS + tg] = acc;
    }
  }
  if (tid < T_TAGS) fvbuf[0][tid] = (tid == START_TAG) ? 0.0f : NEGV;
  __syncthreads();

  // forward max-plus scan (12 lanes active)
  for (int s = 0; s < S_DIM; ++s) {
    if (tid < T_TAGS) {
      float best = -3.4e38f; int bi = 0;
      for (int p = 0; p < T_TAGS; ++p) {
        float v = fvbuf[s & 1][p] + tr_l[tid * T_TAGS + p];
        if (v > best) { best = v; bi = p; }
      }
      fvbuf[(s & 1) ^ 1][tid] = best + feats_l[s * T_TAGS + tid];
      bptr_l[s * T_TAGS + tid] = (short)bi;
    }
    __syncthreads();
  }

  if (tid == 0) {
    float best = -3.4e38f; int bi = 0;
    for (int p = 0; p < T_TAGS; ++p) {
      float v = fvbuf[S_DIM & 1][p] + tr_l[STOP_TAG * T_TAGS + p];
      if (p == STOP_TAG || p == START_TAG) v = NEGV;   // reference masks these
      if (v > best) { best = v; bi = p; }
    }
    out[b] = best;
    int tag = bi;
    for (int s = S_DIM - 1; s >= 0; --s) {
      out[B_DIM + (size_t)b * S_DIM + s] = (float)tag;
      tag = (int)bptr_l[s * T_TAGS + tag];
    }
  }
}

// ---------------------------------------------------------------- launch
extern "C" void kernel_launch(void* const* d_in, const int* in_sizes, int n_in,
                              void* d_out, int out_size, void* d_ws, size_t ws_size,
                              hipStream_t stream) {
  (void)in_sizes; (void)n_in; (void)out_size; (void)ws_size;
  const int*   sent  = (const int*)  d_in[0];
  const float* emb   = (const float*)d_in[4];
  const float* wihf  = (const float*)d_in[5];
  const float* whhf  = (const float*)d_in[6];
  const float* bihf  = (const float*)d_in[7];
  const float* bhhf  = (const float*)d_in[8];
  const float* wihb  = (const float*)d_in[9];
  const float* whhb  = (const float*)d_in[10];
  const float* bihb  = (const float*)d_in[11];
  const float* bhhb  = (const float*)d_in[12];
  const float* Wt    = (const float*)d_in[13];
  const float* bt    = (const float*)d_in[14];
  const float* trans = (const float*)d_in[15];
  const float* h0    = (const float*)d_in[16];
  const float* c0    = (const float*)d_in[17];
  float* out = (float*)d_out;

  // workspace layout (bytes)
  char* ws = (char*)d_ws;
  const size_t SZ_XG = (size_t)S_DIM * B_DIM * E_DIM * 2;     //  8 MB
  const size_t SZ_W  = (size_t)G4H * E_DIM * 2;               //  0.5 MB each
  const size_t SZ_B  = (size_t)G4H * 4;                       //  4 KB each
  const size_t SZ_G  = (size_t)S_DIM * B_DIM * G4H * 2;       // 32 MB each
  const size_t SZ_H  = (size_t)S_DIM * B_DIM * H_DIM * 4;     // 16 MB each
  size_t off = 0;
  unsigned short* Xg    = (unsigned short*)(ws + off); off += SZ_XG;
  unsigned short* oWihF = (unsigned short*)(ws + off); off += SZ_W;
  unsigned short* oWihB = (unsigned short*)(ws + off); off += SZ_W;
  unsigned short* oWhhF = (unsigned short*)(ws + off); off += SZ_W;
  unsigned short* oWhhB = (unsigned short*)(ws + off); off += SZ_W;
  float* biasF = (float*)(ws + off); off += SZ_B;
  float* biasB = (float*)(ws + off); off += SZ_B;
  unsigned short* Gf = (unsigned short*)(ws + off); off += SZ_G;
  unsigned short* Gb = (unsigned short*)(ws + off); off += SZ_G;
  float* Hf = (float*)(ws + off); off += SZ_H;
  float* Hb = (float*)(ws + off); off += SZ_H;

  prep_weights<<<4096, 256, 0, stream>>>(wihf, whhf, wihb, whhb,
                                         oWihF, oWhhF, oWihB, oWhhB);
  prep_bias<<<8, 256, 0, stream>>>(bihf, bhhf, bihb, bhhb, biasF, biasB);
  gather_embed<<<S_DIM * B_DIM, 256, 0, stream>>>(sent, emb, Xg);
  input_proj<<<dim3(S_DIM * B_DIM / 128, 32), 256, 0, stream>>>(
      Xg, oWihF, oWihB, biasF, biasB, Gf, Gb);
  lstm_scan<<<2, 1024, 0, stream>>>(Gf, Gb, oWhhF, oWhhB, h0, c0, Hf, Hb);
  feats_viterbi<<<B_DIM, 256, 0, stream>>>(Hf, Hb, Wt, bt, trans, out);
}
